// QJoint_86105504350314
// MI455X (gfx1250) — matrix operand order repaired and checked by reading, not compiled, not run on hardware
//
#include <hip/hip_runtime.h>

typedef __attribute__((ext_vector_type(16))) _Float16 v16h;
typedef __attribute__((ext_vector_type(8)))  float    v8f;

#define NW 4   // waves per block (128 threads, wave32)

__device__ __forceinline__ v8f vzero() {
  v8f z = {0.f,0.f,0.f,0.f,0.f,0.f,0.f,0.f};
  return z;
}

__device__ __forceinline__ v8f wmma16(v16h a, v16h b, v8f c) {
  // D = A(16x32 f16) * B(32x16 f16) + C(16x16 f32)
  return __builtin_amdgcn_wmma_f32_16x16x32_f16(false, a, false, b, (short)0, c, false, false);
}

__device__ __forceinline__ float elu1(float x) { return x > 0.f ? x : expm1f(x); }

// ---- operand loaders ---------------------------------------------------
// B operand (32x16): weights stored transposed [Ncols][Kpad]; lane l covers
// column n0+(l&15), K rows k0 + kb+0..7 and k0 + 16+kb+0..7 with kb=(l>=16)?8:0.
__device__ __forceinline__ v16h load_b16(const _Float16* W, int kstride, int k0, int n0, int lane) {
  const _Float16* p = W + (n0 + (lane & 15)) * kstride + k0 + ((lane >> 4) << 3);
  v16h b;
#pragma unroll
  for (int j = 0; j < 8; ++j) { b[j] = p[j]; b[8 + j] = p[16 + j]; }
  return b;
}

// A operand (16x32) from LDS f16 row-major [16][rs]: lane l holds row l&15,
// K = k0 + kb+0..7 and k0 + 16+kb+0..7 with kb=(l>=16)?8:0.
__device__ __forceinline__ v16h load_a16(const _Float16* X, int rs, int k0, int lane) {
  const _Float16* row = X + (lane & 15) * rs;
  int kb = k0 + ((lane >> 4) << 3);
  v16h a;
#pragma unroll
  for (int j = 0; j < 8; ++j) { a[j] = row[kb + j]; a[8 + j] = row[kb + 16 + j]; }
  return a;
}

// A operand from global f32 row-major with row clamp and K zero-pad.
__device__ __forceinline__ v16h load_a_f32(const float* X, long long row0, long long rowlim,
                                           int stride, int k0, int kvalid, int lane) {
  long long r = row0 + (lane & 15);
  if (r >= rowlim) r = rowlim - 1;
  const float* row = X + r * stride;
  int kb = k0 + ((lane >> 4) << 3);
  v16h a;
#pragma unroll
  for (int j = 0; j < 8; ++j) {
    int k1 = kb + j, k2 = kb + 16 + j;
    a[j]     = (k1 < kvalid) ? (_Float16)row[k1] : (_Float16)0.f;
    a[8 + j] = (k2 < kvalid) ? (_Float16)row[k2] : (_Float16)0.f;
  }
  return a;
}

// A operand = segment mean (sum/cnt) rows from ws, f32 -> f16.
__device__ __forceinline__ v16h load_a_mean(const float* sum, const float* cnt,
                                            long long row0, long long G, int k0, int lane) {
  long long gr = row0 + (lane & 15);
  if (gr >= G) gr = G - 1;
  float inv = 1.f / fmaxf(cnt[gr], 1.f);
  const float* row = sum + gr * 128;
  int kb = k0 + ((lane >> 4) << 3);
  v16h a;
#pragma unroll
  for (int j = 0; j < 8; ++j) {
    a[j]     = (_Float16)(row[kb + j] * inv);
    a[8 + j] = (_Float16)(row[kb + 16 + j] * inv);
  }
  return a;
}

// ---- shared pair-MLP3 (130->64->64->128), result left in acc3 (pre-bias3)
__device__ __forceinline__ void mlp_key1(const float* encA, long long row0, long long N,
                                         const _Float16* w1p, const _Float16* w2t, const _Float16* w3t,
                                         const float* b1, const float* b2,
                                         _Float16* H, int lane, v8f acc3[8]) {
  int nn = lane & 15, mb = (lane >> 4) << 3;
  v8f acc[4];
#pragma unroll
  for (int n = 0; n < 4; ++n) acc[n] = vzero();
#pragma unroll
  for (int kc = 0; kc < 5; ++kc) {           // K padded 130 -> 160
    v16h a = load_a_f32(encA, row0, N, 130, kc * 32, 130, lane);
#pragma unroll
    for (int n = 0; n < 4; ++n)
      acc[n] = wmma16(a, load_b16(w1p, 160, kc * 32, n * 16, lane), acc[n]);
  }
#pragma unroll
  for (int n = 0; n < 4; ++n) {              // bias + ELU -> LDS (layout transpose)
    int col = n * 16 + nn; float bb = b1[col];
#pragma unroll
    for (int r = 0; r < 8; ++r) H[(mb + r) * 64 + col] = (_Float16)elu1(acc[n][r] + bb);
  }
  v8f a2[4];
#pragma unroll
  for (int n = 0; n < 4; ++n) a2[n] = vzero();
#pragma unroll
  for (int kc = 0; kc < 2; ++kc) {           // 64 -> 64
    v16h a = load_a16(H, 64, kc * 32, lane);
#pragma unroll
    for (int n = 0; n < 4; ++n)
      a2[n] = wmma16(a, load_b16(w2t, 64, kc * 32, n * 16, lane), a2[n]);
  }
#pragma unroll
  for (int n = 0; n < 4; ++n) {
    int col = n * 16 + nn; float bb = b2[col];
#pragma unroll
    for (int r = 0; r < 8; ++r) H[(mb + r) * 64 + col] = (_Float16)elu1(a2[n][r] + bb);
  }
#pragma unroll
  for (int n = 0; n < 8; ++n) acc3[n] = vzero();
#pragma unroll
  for (int kc = 0; kc < 2; ++kc) {           // 64 -> 128 (linear, bias added by caller)
    v16h a = load_a16(H, 64, kc * 32, lane);
#pragma unroll
    for (int n = 0; n < 8; ++n)
      acc3[n] = wmma16(a, load_b16(w3t, 64, kc * 32, n * 16, lane), acc3[n]);
  }
}

// ---- kernels -----------------------------------------------------------
__global__ void k_zero(float* p, long long n) {
  long long i = (long long)blockIdx.x * blockDim.x + threadIdx.x;
  long long st = (long long)gridDim.x * blockDim.x;
  for (; i < n; i += st) p[i] = 0.f;
}

// convert all GEMM weights f32 -> f16, transposed [N][Kpad] (zero-padded K)
__global__ void k_prep(const float* p1w1, const float* p1w2, const float* p1w3,
                       const float* gw1, const float* gw2, const float* p2w1,
                       _Float16* w1p, _Float16* w2t, _Float16* w3t,
                       _Float16* gw1t, _Float16* gw2t, _Float16* p2w1t) {
  int i = blockIdx.x * blockDim.x + threadIdx.x;
  if (i < 64 * 160) { int n = i / 160, k = i % 160;
    w1p[i] = (k < 130) ? (_Float16)p1w1[k * 64 + n] : (_Float16)0.f; return; }
  i -= 64 * 160;
  if (i < 64 * 64)  { int n = i / 64,  k = i % 64;  w2t[i]   = (_Float16)p1w2[k * 64 + n];  return; }
  i -= 64 * 64;
  if (i < 128 * 64) { int n = i / 64,  k = i % 64;  w3t[i]   = (_Float16)p1w3[k * 128 + n]; return; }
  i -= 128 * 64;
  if (i < 64 * 128) { int n = i / 128, k = i % 128; gw1t[i]  = (_Float16)gw1[k * 64 + n];   return; }
  i -= 64 * 128;
  if (i < 64 * 64)  { int n = i / 64,  k = i % 64;  gw2t[i]  = (_Float16)gw2[k * 64 + n];   return; }
  i -= 64 * 64;
  if (i < 64 * 128) { int n = i / 128, k = i % 128; p2w1t[i] = (_Float16)p2w1[k * 64 + n];  return; }
}

// phase 1: key1 = MLP3(encA); atomically accumulate segment sums + counts
__global__ void __launch_bounds__(32 * NW)
k_phase1(const float* encA, const int* gidx,
         const _Float16* w1p, const _Float16* w2t, const _Float16* w3t,
         const float* b1, const float* b2, const float* b3,
         float* gsum, float* gcnt, long long N, int ptiles) {
  __shared__ _Float16 sH[NW][16 * 64];
  int wave = threadIdx.x >> 5, lane = threadIdx.x & 31;
  int tile = blockIdx.x * NW + wave;
  if (tile >= ptiles) return;
  long long row0 = (long long)tile * 16;
  v8f acc3[8];
  mlp_key1(encA, row0, N, w1p, w2t, w3t, b1, b2, sH[wave], lane, acc3);
  int nn = lane & 15, mb = (lane >> 4) << 3;
#pragma unroll
  for (int n = 0; n < 8; ++n) {
    int col = n * 16 + nn; float bb = b3[col];
#pragma unroll
    for (int r = 0; r < 8; ++r) {
      long long row = row0 + mb + r;
      if (row < N) {
        int g = gidx[row];
        atomicAdd(&gsum[(long long)g * 128 + col], acc3[n][r] + bb);
      }
    }
  }
  if (lane < 16 && (row0 + lane) < N) atomicAdd(&gcnt[gidx[row0 + lane]], 1.0f);
}

// group MLP3: q_jt = MLP(mean) -> out_q[G]
__global__ void __launch_bounds__(32 * NW)
k_group(const float* gsum, const float* gcnt,
        const _Float16* gw1t, const _Float16* gw2t,
        const float* gb1, const float* gb2, const float* gw3, const float* gb3,
        float* out_q, long long G, int gtiles) {
  __shared__ _Float16 sH[NW][16 * 64];
  int wave = threadIdx.x >> 5, lane = threadIdx.x & 31;
  int tile = blockIdx.x * NW + wave;
  if (tile >= gtiles) return;
  long long row0 = (long long)tile * 16;
  _Float16* H = sH[wave];
  int nn = lane & 15, mb = (lane >> 4) << 3;
  v8f acc[4];
#pragma unroll
  for (int n = 0; n < 4; ++n) acc[n] = vzero();
#pragma unroll
  for (int kc = 0; kc < 4; ++kc) {           // 128 -> 64
    v16h a = load_a_mean(gsum, gcnt, row0, G, kc * 32, lane);
#pragma unroll
    for (int n = 0; n < 4; ++n)
      acc[n] = wmma16(a, load_b16(gw1t, 128, kc * 32, n * 16, lane), acc[n]);
  }
#pragma unroll
  for (int n = 0; n < 4; ++n) {
    int col = n * 16 + nn; float bb = gb1[col];
#pragma unroll
    for (int r = 0; r < 8; ++r) H[(mb + r) * 64 + col] = (_Float16)elu1(acc[n][r] + bb);
  }
  v8f a2[4];
#pragma unroll
  for (int n = 0; n < 4; ++n) a2[n] = vzero();
#pragma unroll
  for (int kc = 0; kc < 2; ++kc) {           // 64 -> 64
    v16h a = load_a16(H, 64, kc * 32, lane);
#pragma unroll
    for (int n = 0; n < 4; ++n)
      a2[n] = wmma16(a, load_b16(gw2t, 64, kc * 32, n * 16, lane), a2[n]);
  }
#pragma unroll
  for (int n = 0; n < 4; ++n) {
    int col = n * 16 + nn; float bb = gb2[col];
#pragma unroll
    for (int r = 0; r < 8; ++r) H[(mb + r) * 64 + col] = (_Float16)elu1(a2[n][r] + bb);
  }
  if (lane < 16) {                            // 64 -> 1 head (VALU dot)
    long long gr = row0 + lane;
    if (gr < G) {
      float s = gb3[0];
#pragma unroll
      for (int k = 0; k < 64; ++k) s += (float)H[lane * 64 + k] * gw3[k];
      out_q[gr] = s;
    }
  }
}

// phase 2: recompute key1, alt = enc + mean[g] - key1, alt_q = MLP2(alt)
__global__ void __launch_bounds__(32 * NW)
k_phase2(const float* enc, const float* encA, const int* gidx,
         const _Float16* w1p, const _Float16* w2t, const _Float16* w3t,
         const float* b1, const float* b2, const float* b3,
         const float* gsum, const float* gcnt,
         const _Float16* p2w1t, const float* p2b1, const float* p2w2, const float* p2b2,
         float* out_alt, long long N, int ptiles) {
  __shared__ _Float16 sA[NW][16 * 128];  // alt tile (A-side of p2 GEMM)
  __shared__ _Float16 sH[NW][16 * 64];
  int wave = threadIdx.x >> 5, lane = threadIdx.x & 31;
  int tile = blockIdx.x * NW + wave;
  if (tile >= ptiles) return;
  long long row0 = (long long)tile * 16;
  v8f acc3[8];
  mlp_key1(encA, row0, N, w1p, w2t, w3t, b1, b2, sH[wave], lane, acc3);

  int nn = lane & 15, mb = (lane >> 4) << 3;
  int g[8]; float inv[8];
#pragma unroll
  for (int r = 0; r < 8; ++r) {
    long long row = row0 + mb + r;
    if (row >= N) row = N - 1;
    g[r] = gidx[row];
    inv[r] = 1.f / fmaxf(gcnt[g[r]], 1.f);
  }
  _Float16* A = sA[wave];
#pragma unroll
  for (int n = 0; n < 8; ++n) {
    int col = n * 16 + nn; float bb = b3[col];
#pragma unroll
    for (int r = 0; r < 8; ++r) {
      long long row = row0 + mb + r;
      long long rc = (row < N) ? row : (N - 1);
      float key1v = acc3[n][r] + bb;
      float alt = enc[rc * 128 + col] + gsum[(long long)g[r] * 128 + col] * inv[r] - key1v;
      A[(mb + r) * 128 + col] = (_Float16)alt;
    }
  }
  v8f a4[4];
#pragma unroll
  for (int n = 0; n < 4; ++n) a4[n] = vzero();
#pragma unroll
  for (int kc = 0; kc < 4; ++kc) {           // 128 -> 64
    v16h a = load_a16(A, 128, kc * 32, lane);
#pragma unroll
    for (int n = 0; n < 4; ++n)
      a4[n] = wmma16(a, load_b16(p2w1t, 128, kc * 32, n * 16, lane), a4[n]);
  }
  _Float16* H = sH[wave];
#pragma unroll
  for (int n = 0; n < 4; ++n) {
    int col = n * 16 + nn; float bb = p2b1[col];
#pragma unroll
    for (int r = 0; r < 8; ++r) H[(mb + r) * 64 + col] = (_Float16)elu1(a4[n][r] + bb);
  }
  {                                          // 64 -> 2 head: lane covers (row, oc)
    int row = lane & 15, oc = lane >> 4;
    long long grow = row0 + row;
    if (grow < N) {
      float s = p2b2[oc];
#pragma unroll
      for (int k = 0; k < 64; ++k) s += (float)H[row * 64 + k] * p2w2[k * 2 + oc];
      out_alt[grow * 2 + oc] = s;
    }
  }
}

// ---- launch ------------------------------------------------------------
extern "C" void kernel_launch(void* const* d_in, const int* in_sizes, int n_in,
                              void* d_out, int out_size, void* d_ws, size_t ws_size,
                              hipStream_t stream) {
  const float* enc  = (const float*)d_in[0];
  const float* encA = (const float*)d_in[1];
  const int*   gidx = (const int*)d_in[2];
  const float* p1w1 = (const float*)d_in[4];
  const float* p1b1 = (const float*)d_in[5];
  const float* p1w2 = (const float*)d_in[6];
  const float* p1b2 = (const float*)d_in[7];
  const float* p1w3 = (const float*)d_in[8];
  const float* p1b3 = (const float*)d_in[9];
  const float* gw1  = (const float*)d_in[10];
  const float* gb1  = (const float*)d_in[11];
  const float* gw2  = (const float*)d_in[12];
  const float* gb2  = (const float*)d_in[13];
  const float* gw3  = (const float*)d_in[14];
  const float* gb3  = (const float*)d_in[15];
  const float* p2w1 = (const float*)d_in[16];
  const float* p2b1 = (const float*)d_in[17];
  const float* p2w2 = (const float*)d_in[18];
  const float* p2b2 = (const float*)d_in[19];

  long long N = (long long)in_sizes[0] / 128;        // 500000
  long long G = (long long)out_size - 2 * N;         // 31250 (out = G + 2N)

  float* gsum = (float*)d_ws;                        // [G][128]
  float* gcnt = gsum + (size_t)G * 128;              // [G]
  size_t woff = ((size_t)G * 129 * sizeof(float) + 255) & ~(size_t)255;
  _Float16* w1p   = (_Float16*)((char*)d_ws + woff); // [64][160]
  _Float16* w2t   = w1p   + 64 * 160;                // [64][64]
  _Float16* w3t   = w2t   + 64 * 64;                 // [128][64]
  _Float16* gw1t  = w3t   + 128 * 64;                // [64][128]
  _Float16* gw2t  = gw1t  + 64 * 128;                // [64][64]
  _Float16* p2w1t = gw2t  + 64 * 64;                 // [64][128]

  float* out_q   = (float*)d_out;                    // [G]
  float* out_alt = out_q + G;                        // [N][2]

  int ptiles = (int)((N + 15) / 16);
  int gtiles = (int)((G + 15) / 16);
  int pblocks = (ptiles + NW - 1) / NW;
  int gblocks = (gtiles + NW - 1) / NW;

  k_zero<<<4096, 256, 0, stream>>>(gsum, G * 129);
  k_prep<<<(43008 + 255) / 256, 256, 0, stream>>>(p1w1, p1w2, p1w3, gw1, gw2, p2w1,
                                                  w1p, w2t, w3t, gw1t, gw2t, p2w1t);
  k_phase1<<<pblocks, 32 * NW, 0, stream>>>(encA, gidx, w1p, w2t, w3t,
                                            p1b1, p1b2, p1b3, gsum, gcnt, N, ptiles);
  k_group<<<gblocks, 32 * NW, 0, stream>>>(gsum, gcnt, gw1t, gw2t,
                                           gb1, gb2, gw3, gb3, out_q, G, gtiles);
  k_phase2<<<pblocks, 32 * NW, 0, stream>>>(enc, encA, gidx, w1p, w2t, w3t,
                                            p1b1, p1b2, p1b3, gsum, gcnt,
                                            p2w1t, p2b1, p2w2, p2b2, out_alt, N, ptiles);
}
